// RecurrentRGCN_37297495998618
// MI455X (gfx1250) — compile-verified, matrix operand order checked by reading, not run on hardware
//
#include <hip/hip_runtime.h>
#include <hip/hip_bf16.h>
#include <math.h>

typedef __attribute__((ext_vector_type(16))) __bf16 v16bf;
typedef __attribute__((ext_vector_type(8)))  __bf16 v8bf;
typedef __attribute__((ext_vector_type(8)))  float  v8f;

#define RRELU_SLOPE 0.22916666666666666f  // (1/8 + 1/3)/2

__device__ __forceinline__ unsigned short f2bf(float f) {
  unsigned u = __builtin_bit_cast(unsigned, f);
  u = u + 0x7FFFu + ((u >> 16) & 1u);   // round-to-nearest-even
  return (unsigned short)(u >> 16);
}

// ---------------------------------------------------------------------------
// C[M,Ncols] = A[M,K] @ B[K,Ncols].  A is bf16 [M,K]; Bt = B^T bf16 [Ncols,K].
// One wave computes a 16x64 C strip: per K-step the A fragment (2 x b128) is
// reused across 4 v_wmma_f32_16x16x32_bf16.  K is a template constant so the
// whole reduction is unrolled into clause-able straight-line loads.
// Ragged M handled by clamping the A row (dup row; its outputs are never
// stored), keeping EXEC all-ones and the loop body branch-free.
// Requires: K % 32 == 0, Ncols % 64 == 0.
// ---------------------------------------------------------------------------
template <int K>
__global__ __launch_bounds__(128) void gemm_bf16_wmma(
    const __bf16* __restrict__ A, const __bf16* __restrict__ Bt,
    float* __restrict__ C, int M, int Ncols)
{
  const int lane = threadIdx.x & 31;
  const int wave = threadIdx.x >> 5;
  const int n0   = (blockIdx.x * 4 + wave) * 64;   // first of 4 N-tiles
  const int tileM = blockIdx.y;
  if (n0 >= Ncols) return;                         // uniform across the wave
  const int half = lane >> 4;                      // 0: K {0..7,16..23}; 1: +8
  const int l15  = lane & 15;
  int rowA = tileM * 16 + l15;
  if (rowA >= M) rowA = M - 1;                     // dup row, outputs discarded
  const __bf16* arow = A + (long long)rowA * K + half * 8;
  const __bf16* brow = Bt + (long long)(n0 + l15) * K + half * 8;

  v8f acc[4] = {};
#pragma unroll
  for (int k0 = 0; k0 < K; k0 += 32) {
    v16bf a = __builtin_shufflevector(
        *(const v8bf*)(arow + k0), *(const v8bf*)(arow + k0 + 16),
        0, 1, 2, 3, 4, 5, 6, 7, 8, 9, 10, 11, 12, 13, 14, 15);
#pragma unroll
    for (int j = 0; j < 4; ++j) {
      const __bf16* bp = brow + j * 16 * K + k0;
      v16bf b = __builtin_shufflevector(
          *(const v8bf*)(bp), *(const v8bf*)(bp + 16),
          0, 1, 2, 3, 4, 5, 6, 7, 8, 9, 10, 11, 12, 13, 14, 15);
      acc[j] = __builtin_amdgcn_wmma_f32_16x16x32_bf16(
          false, a, false, b, (short)0, acc[j], false, false);
    }
  }
#pragma unroll
  for (int j = 0; j < 4; ++j) {
    const int col = n0 + j * 16 + l15;
#pragma unroll
    for (int v = 0; v < 8; ++v) {
      int r = tileM * 16 + half * 8 + v;   // C/D layout: VGPR v -> row v(+8)
      if (r < M) C[(long long)r * Ncols + col] = acc[j][v];
    }
  }
}

// ---------------------------------------------------------------------------
// Weight conversion helpers (f32 -> bf16, optionally transposed)
// ---------------------------------------------------------------------------
__global__ void cvt_bf(const float* __restrict__ s, unsigned short* __restrict__ d, int n) {
  int i = blockIdx.x * blockDim.x + threadIdx.x;
  if (i < n) d[i] = f2bf(s[i]);
}
__global__ void cvt_bf_t(const float* __restrict__ s, unsigned short* __restrict__ d,
                         int rows, int cols) {
  // d[c*rows + r] = s[r*cols + c]   (d is [cols,rows] row-major)
  int i = blockIdx.x * blockDim.x + threadIdx.x;
  if (i < rows * cols) {
    int c = i / rows, r = i % rows;
    d[i] = f2bf(s[(long long)r * cols + c]);
  }
}
__global__ void copy_f(const float* __restrict__ s, float* __restrict__ d, int n) {
  int i = blockIdx.x * blockDim.x + threadIdx.x;
  if (i < n) d[i] = s[i];
}

// ---------------------------------------------------------------------------
// Row-wise L2 normalize (H == 256 == blockDim.x), LDS tree reduction.
// Emits f32 result + bf16 mirror (GEMM operand).
// ---------------------------------------------------------------------------
__global__ __launch_bounds__(256) void row_norm_copy(
    const float* __restrict__ src, float* __restrict__ dst,
    unsigned short* __restrict__ dst_bf) {
  __shared__ float red[256];
  long long row = blockIdx.x;
  int j = threadIdx.x;
  float v = src[row * 256 + j];
  red[j] = v * v; __syncthreads();
  for (int s = 128; s > 0; s >>= 1) { if (j < s) red[j] += red[j + s]; __syncthreads(); }
  float out = v / fmaxf(sqrtf(red[0]), 1e-12f);
  dst[row * 256 + j] = out;
  dst_bf[row * 256 + j] = f2bf(out);
}

// ---------------------------------------------------------------------------
// Relation segment sum: per incidence l, sums[rid[l]] += h[rte[l]], cnt++
// One thread per (l, 4-float chunk); h/sums L2-resident, f32 global atomics.
// ---------------------------------------------------------------------------
__global__ void rel_gather_sum(const int* __restrict__ rte, const int* __restrict__ rid,
                               const float* __restrict__ h, float* sums, float* cnt,
                               long long total) {
  long long idx = (long long)blockIdx.x * blockDim.x + threadIdx.x;
  if (idx >= total) return;
  int l = (int)(idx >> 6);
  int c4 = (int)(idx & 63);
  int e = rte[l], r = rid[l];
  const float4 v = *(const float4*)(h + (long long)e * 256 + c4 * 4);
  float* s = sums + (long long)r * 256 + c4 * 4;
  atomicAdd(s + 0, v.x); atomicAdd(s + 1, v.y);
  atomicAdd(s + 2, v.z); atomicAdd(s + 3, v.w);
  if (c4 == 0) atomicAdd(cnt + r, 1.0f);
}

// x_cat = [emb_rel | sums/max(cnt,1)] written directly in bf16 (GEMM operand)
__global__ void build_xcat(const float* __restrict__ emb_rel, const float* __restrict__ sums,
                           const float* __restrict__ cnt, unsigned short* __restrict__ xcat,
                           int n) {
  int i = blockIdx.x * blockDim.x + threadIdx.x;
  if (i >= n) return;
  int r = i >> 9;          // row (2H = 512 wide)
  int c = i & 511;
  float v = (c < 256) ? emb_rel[r * 256 + c]
                      : sums[r * 256 + (c - 256)] / fmaxf(cnt[r], 1.0f);
  xcat[i] = f2bf(v);
}

// ---------------------------------------------------------------------------
// GRU gate combine + L2 normalize; updates hstate (f32) + hstate_bf (bf16)
// in place (row-private); optionally mirrors into out2 (T-history output).
// ---------------------------------------------------------------------------
__global__ __launch_bounds__(256) void gru_norm(
    const float* __restrict__ gi, const float* __restrict__ gh,
    const float* __restrict__ bih, const float* __restrict__ bhh,
    float* hstate, unsigned short* hstate_bf, float* out2) {
  __shared__ float red[256];
  long long row = blockIdx.x;
  int j = threadIdx.x;
  const float* gir = gi + row * 768;
  const float* ghr = gh + row * 768;
  float ir = gir[j]       + bih[j];
  float iz = gir[256 + j] + bih[256 + j];
  float in = gir[512 + j] + bih[512 + j];
  float hr = ghr[j]       + bhh[j];
  float hz = ghr[256 + j] + bhh[256 + j];
  float hn = ghr[512 + j] + bhh[512 + j];
  float r = 1.0f / (1.0f + expf(-(ir + hr)));
  float z = 1.0f / (1.0f + expf(-(iz + hz)));
  float n = tanhf(in + r * hn);
  float v = (1.0f - z) * n + z * hstate[row * 256 + j];
  red[j] = v * v; __syncthreads();
  for (int s = 128; s > 0; s >>= 1) { if (j < s) red[j] += red[j + s]; __syncthreads(); }
  v /= fmaxf(sqrtf(red[0]), 1e-12f);
  hstate[row * 256 + j] = v;
  hstate_bf[row * 256 + j] = f2bf(v);
  if (out2) out2[row * 256 + j] = v;
}

// ---------------------------------------------------------------------------
// Edge scatter: agg[dst] += hW[src] + h0W[etype]; deg[dst] += 1
// (msg GEMM pre-distributed:  (h[s]+h0[et])@W = (h@W)[s] + (h0@W)[et])
// ---------------------------------------------------------------------------
__global__ void edge_scatter(const int* __restrict__ srcI, const int* __restrict__ dstI,
                             const int* __restrict__ etI, const float* __restrict__ hW,
                             const float* __restrict__ h0W, float* agg, float* deg,
                             long long total) {
  long long idx = (long long)blockIdx.x * blockDim.x + threadIdx.x;
  if (idx >= total) return;
  int e = (int)(idx >> 6);
  int c4 = (int)(idx & 63);
  int s = srcI[e], d = dstI[e], t = etI[e];
  const float4 a = *(const float4*)(hW + (long long)s * 256 + c4 * 4);
  const float4 b = *(const float4*)(h0W + (long long)t * 256 + c4 * 4);
  float* o = agg + (long long)d * 256 + c4 * 4;
  atomicAdd(o + 0, a.x + b.x); atomicAdd(o + 1, a.y + b.y);
  atomicAdd(o + 2, a.z + b.z); atomicAdd(o + 3, a.w + b.w);
  if (c4 == 0) atomicAdd(deg + d, 1.0f);
}

// cur = rrelu(agg/max(deg,1) + h@W_loop), row-normalized; bf16 out only
// (cur is consumed solely as a GEMM A operand)
__global__ __launch_bounds__(256) void cur_norm(const float* __restrict__ agg,
                                                const float* __restrict__ deg,
                                                const float* __restrict__ hL,
                                                unsigned short* __restrict__ cur_bf) {
  __shared__ float red[256];
  long long row = blockIdx.x;
  int j = threadIdx.x;
  float v = agg[row * 256 + j] / fmaxf(deg[row], 1.0f) + hL[row * 256 + j];
  v = (v >= 0.0f) ? v : RRELU_SLOPE * v;
  red[j] = v * v; __syncthreads();
  for (int s = 128; s > 0; s >>= 1) { if (j < s) red[j] += red[j + s]; __syncthreads(); }
  float scale = 1.0f / fmaxf(sqrtf(red[0]), 1e-12f);
  cur_bf[row * 256 + j] = f2bf(v * scale);
}

// ---------------------------------------------------------------------------
extern "C" void kernel_launch(void* const* d_in, const int* in_sizes, int n_in,
                              void* d_out, int out_size, void* d_ws, size_t ws_size,
                              hipStream_t stream) {
  (void)in_sizes; (void)n_in; (void)out_size; (void)ws_size;
  const int N = 20000, H = 256, R2 = 500, T = 8, E = 200000, Lc = 400000;
  const int H2 = 512, H3 = 768;

  const int*   r_to_e  = (const int*)d_in[0];
  const int*   rel_ids = (const int*)d_in[1];
  const int*   srcI    = (const int*)d_in[2];
  const int*   dstI    = (const int*)d_in[3];
  const int*   etI     = (const int*)d_in[4];
  const float* dyn_emb = (const float*)d_in[5];
  const float* emb_rel = (const float*)d_in[6];
  const float* W_ih_rel = (const float*)d_in[7];
  const float* W_hh_rel = (const float*)d_in[8];
  const float* b_ih_rel = (const float*)d_in[9];
  const float* b_hh_rel = (const float*)d_in[10];
  const float* W_msg    = (const float*)d_in[11];
  const float* W_loop   = (const float*)d_in[12];
  const float* W_ih_ent = (const float*)d_in[13];
  const float* W_hh_ent = (const float*)d_in[14];
  const float* b_ih_ent = (const float*)d_in[15];
  const float* b_hh_ent = (const float*)d_in[16];
  float* hist = (float*)d_out;

  char* p = (char*)d_ws;
  auto allocF = [&](size_t n) -> float* {
    float* r = (float*)p; p += ((n * 4 + 255) / 256) * 256; return r; };
  auto allocU = [&](size_t n) -> unsigned short* {
    unsigned short* r = (unsigned short*)p; p += ((n * 2 + 255) / 256) * 256; return r; };

  float* h    = allocF((size_t)N * H);
  float* h0   = allocF((size_t)R2 * H);
  float* sums = allocF((size_t)R2 * H);
  float* cnt  = allocF(R2);
  float* gi_r = allocF((size_t)R2 * H3);
  float* gh_r = allocF((size_t)R2 * H3);
  float* hW   = allocF((size_t)N * H);
  float* h0W  = allocF((size_t)R2 * H);
  float* hL   = allocF((size_t)N * H);
  float* agg  = allocF((size_t)N * H);
  float* deg  = allocF(N);
  float* gi_e = allocF((size_t)N * H3);
  float* gh_e = allocF((size_t)N * H3);
  unsigned short* h_bf   = allocU((size_t)N * H);
  unsigned short* h0_bf  = allocU((size_t)R2 * H);
  unsigned short* cur_bf = allocU((size_t)N * H);
  unsigned short* xcat_bf= allocU((size_t)R2 * H2);
  unsigned short* bWihR  = allocU((size_t)H3 * H2);
  unsigned short* bWhhR  = allocU((size_t)H3 * H);
  unsigned short* bWihE  = allocU((size_t)H3 * H);
  unsigned short* bWhhE  = allocU((size_t)H3 * H);
  unsigned short* bWmsgT = allocU((size_t)H * H);
  unsigned short* bWloopT= allocU((size_t)H * H);

  const int th = 256;
  // bf16 weight copies.  gi = x @ W^T needs B^T = W (row-major, as-is);
  // x @ W_msg / x @ W_loop need explicit transposes.
  cvt_bf  <<<(H3 * H2 + th - 1) / th, th, 0, stream>>>(W_ih_rel, bWihR, H3 * H2);
  cvt_bf  <<<(H3 * H  + th - 1) / th, th, 0, stream>>>(W_hh_rel, bWhhR, H3 * H);
  cvt_bf  <<<(H3 * H  + th - 1) / th, th, 0, stream>>>(W_ih_ent, bWihE, H3 * H);
  cvt_bf  <<<(H3 * H  + th - 1) / th, th, 0, stream>>>(W_hh_ent, bWhhE, H3 * H);
  cvt_bf_t<<<(H * H   + th - 1) / th, th, 0, stream>>>(W_msg,  bWmsgT,  H, H);
  cvt_bf_t<<<(H * H   + th - 1) / th, th, 0, stream>>>(W_loop, bWloopT, H, H);

  row_norm_copy<<<N, 256, 0, stream>>>(dyn_emb, h, h_bf);
  copy_f<<<(R2 * H + th - 1) / th, th, 0, stream>>>(emb_rel, h0, R2 * H);
  cvt_bf<<<(R2 * H + th - 1) / th, th, 0, stream>>>(emb_rel, h0_bf, R2 * H);

  for (int t = 0; t < T; ++t) {
    const int* rte = r_to_e  + (size_t)t * Lc;
    const int* rid = rel_ids + (size_t)t * Lc;
    const int* s_  = srcI + (size_t)t * E;
    const int* d_  = dstI + (size_t)t * E;
    const int* et  = etI  + (size_t)t * E;

    hipMemsetAsync(sums, 0, (size_t)R2 * H * 4, stream);
    hipMemsetAsync(cnt,  0, (size_t)R2 * 4, stream);
    {
      long long tot = (long long)Lc * 64;
      rel_gather_sum<<<(unsigned)((tot + 255) / 256), 256, 0, stream>>>(
          rte, rid, h, sums, cnt, tot);
    }
    build_xcat<<<(R2 * H2 + th - 1) / th, th, 0, stream>>>(emb_rel, sums, cnt,
                                                           xcat_bf, R2 * H2);

    { dim3 g((H3 + 255) / 256, (R2 + 15) / 16);
      gemm_bf16_wmma<512><<<g, 128, 0, stream>>>((const __bf16*)xcat_bf,
                                                 (const __bf16*)bWihR, gi_r, R2, H3); }
    { dim3 g((H3 + 255) / 256, (R2 + 15) / 16);
      gemm_bf16_wmma<256><<<g, 128, 0, stream>>>((const __bf16*)h0_bf,
                                                 (const __bf16*)bWhhR, gh_r, R2, H3); }
    gru_norm<<<R2, 256, 0, stream>>>(gi_r, gh_r, b_ih_rel, b_hh_rel, h0, h0_bf,
                                     (float*)nullptr);

    { dim3 g((H + 255) / 256, (R2 + 15) / 16);
      gemm_bf16_wmma<256><<<g, 128, 0, stream>>>((const __bf16*)h0_bf,
                                                 (const __bf16*)bWmsgT, h0W, R2, H); }
    { dim3 g((H + 255) / 256, N / 16);
      gemm_bf16_wmma<256><<<g, 128, 0, stream>>>((const __bf16*)h_bf,
                                                 (const __bf16*)bWmsgT, hW, N, H); }
    { dim3 g((H + 255) / 256, N / 16);
      gemm_bf16_wmma<256><<<g, 128, 0, stream>>>((const __bf16*)h_bf,
                                                 (const __bf16*)bWloopT, hL, N, H); }

    hipMemsetAsync(agg, 0, (size_t)N * H * 4, stream);
    hipMemsetAsync(deg, 0, (size_t)N * 4, stream);
    {
      long long tot = (long long)E * 64;
      edge_scatter<<<(unsigned)((tot + 255) / 256), 256, 0, stream>>>(
          s_, d_, et, hW, h0W, agg, deg, tot);
    }
    cur_norm<<<N, 256, 0, stream>>>(agg, deg, hL, cur_bf);

    { dim3 g((H3 + 255) / 256, N / 16);
      gemm_bf16_wmma<256><<<g, 128, 0, stream>>>((const __bf16*)cur_bf,
                                                 (const __bf16*)bWihE, gi_e, N, H3); }
    { dim3 g((H3 + 255) / 256, N / 16);
      gemm_bf16_wmma<256><<<g, 128, 0, stream>>>((const __bf16*)h_bf,
                                                 (const __bf16*)bWhhE, gh_e, N, H3); }
    gru_norm<<<N, 256, 0, stream>>>(gi_e, gh_e, b_ih_ent, b_hh_ent, h, h_bf,
                                    hist + (size_t)t * N * H);
  }
}
